// Block_71665824301263
// MI455X (gfx1250) — matrix operand order repaired and checked by reading, not compiled, not run on hardware
//
#include <hip/hip_runtime.h>
#include <math.h>

typedef unsigned short u16;
typedef unsigned int   u32;
typedef unsigned long long u64;

typedef __attribute__((ext_vector_type(16))) __bf16 bf16x16;
typedef __attribute__((ext_vector_type(8)))  float  f32x8;

// ---------- gfx1250 async global->LDS path (guarded) ----------
#if defined(__has_builtin)
#if __has_builtin(__builtin_amdgcn_global_load_async_to_lds_b128)
#define HAVE_ASYNC_LDS 1
#endif
#endif

#if defined(HAVE_ASYNC_LDS)
// builtin expects: (v4i __device__* src, v4i __shared__* dst, imm offset, imm cpol)
// where v4i is a GCC-style vector: __attribute__((vector_size(16))) int
typedef int v4i_ls __attribute__((vector_size(16)));
typedef __attribute__((address_space(1))) v4i_ls* as1_v4i;
typedef __attribute__((address_space(3))) v4i_ls* as3_v4i;
#endif

__device__ __forceinline__ void copy16_to_lds(u16* lds, const u16* g) {
#if defined(HAVE_ASYNC_LDS)
  // int round-trips avoid cross-AS pointer casts; LDS offset = low 32 bits of flat addr
  __builtin_amdgcn_global_load_async_to_lds_b128(
      (as1_v4i)(u64)g, (as3_v4i)(u32)(u64)lds, 0, 0);
#else
  *(uint4*)lds = *(const uint4*)g;
#endif
}

template <int N>
__device__ __forceinline__ void wait_async() {
#if defined(HAVE_ASYNC_LDS)
#if __has_builtin(__builtin_amdgcn_s_wait_asynccnt)
  __builtin_amdgcn_s_wait_asynccnt(N);
#else
  asm volatile("s_wait_asynccnt %0" ::"i"(N) : "memory");
#endif
#endif
}

// ---------- small helpers ----------
__device__ __forceinline__ u16 f2bf(float f) {
  u32 u = __float_as_uint(f);
  u32 r = u + 0x7fffu + ((u >> 16) & 1u);   // round-to-nearest-even
  return (u16)(r >> 16);
}
__device__ __forceinline__ float bf2f(u16 h) {
  return __uint_as_float(((u32)h) << 16);
}
// order-preserving float<->uint map for atomicMax on floats
__device__ __forceinline__ u32 f2ord(float f) {
  u32 u = __float_as_uint(f);
  return (u & 0x80000000u) ? ~u : (u | 0x80000000u);
}
__device__ __forceinline__ float ord2f(u32 e) {
  u32 u = (e & 0x80000000u) ? (e & 0x7fffffffu) : ~e;
  return __uint_as_float(u);
}

// ---------- utility kernels ----------
__global__ void k_zero(u32* __restrict__ p, long n) {
  long i = (long)blockIdx.x * blockDim.x + threadIdx.x;
  if (i < n) p[i] = 0u;
}

__global__ void k_f2bf(const float* __restrict__ in, u16* __restrict__ out, long n) {
  long i = (long)blockIdx.x * blockDim.x + threadIdx.x;
  if (i < n) out[i] = f2bf(in[i]);
}

// W[K][N] fp32 -> Wt[N][K] bf16  (N-major so GEMM B K-pairs are contiguous)
__global__ void k_wtrans(const float* __restrict__ W, u16* __restrict__ Wt, int K, int N) {
  long i = (long)blockIdx.x * blockDim.x + threadIdx.x;
  if (i >= (long)K * N) return;
  int k = (int)(i / N), n = (int)(i % N);
  Wt[(size_t)n * K + k] = f2bf(W[i]);
}

// ---------- batchnorm (column stats over N rows) ----------
__global__ void k_bn_stats(const float* __restrict__ x, float* __restrict__ sum,
                           float* __restrict__ sumsq, int Nrows, int D) {
  int col = threadIdx.x;                 // blockDim.x == D
  int r0 = blockIdx.x * 256;
  int r1 = r0 + 256; if (r1 > Nrows) r1 = Nrows;
  float s = 0.f, ss = 0.f;
  for (int r = r0; r < r1; ++r) {
    float v = x[(size_t)r * D + col];
    s += v; ss += v * v;
  }
  atomicAdd(&sum[col], s);
  atomicAdd(&sumsq[col], ss);
}

__global__ void k_bn_finalize(const float* __restrict__ sum, const float* __restrict__ sumsq,
                              const float* __restrict__ g, const float* __restrict__ b,
                              float* __restrict__ scale, float* __restrict__ shift,
                              int Nrows, int D) {
  int c = blockIdx.x * blockDim.x + threadIdx.x;
  if (c >= D) return;
  float inv = 1.0f / (float)Nrows;
  float mu  = sum[c] * inv;
  float var = sumsq[c] * inv - mu * mu;   // biased var (matches reference)
  float sc  = g[c] * rsqrtf(var + 1e-5f);
  scale[c] = sc;
  shift[c] = b[c] - mu * sc;
}

__global__ void k_bn_apply(const float* __restrict__ x, const float* __restrict__ scale,
                           const float* __restrict__ shift, u16* __restrict__ out,
                           long total, int D) {
  long i = (long)blockIdx.x * blockDim.x + threadIdx.x;
  if (i >= total) return;
  int c = (int)(i % D);
  out[i] = f2bf(fmaf(x[i], scale[c], shift[c]));
}

// ---------- BF16 WMMA GEMM: C[M,N] = A[M,K] @ Wt[N,K]^T (+bias)(+gelu)(+res) ----------
// block tile 128x128, 8 waves (4x2), wave tile 32x64 = 2x4 WMMA 16x16x32 tiles
// Double-buffered LDS; tiles staged with async global->LDS (ASYNCcnt) when available.
union Frag { uint4 q[2]; bf16x16 v; };

__global__ __launch_bounds__(256) void k_gemm(
    const u16* __restrict__ A, const u16* __restrict__ Bt,
    const float* __restrict__ bias, const float* __restrict__ res,
    float* __restrict__ Cf, u16* __restrict__ Cbf,
    int M, int N, int K, int gelu) {
  __shared__ __align__(16) u16 As[2][128 * 32];
  __shared__ __align__(16) u16 Bs[2][128 * 32];

  const int tid  = threadIdx.x;
  const int wave = tid >> 5, lane = tid & 31;
  const int lr = lane & 15, lh = lane >> 4;
  const int wm = wave & 3,  wn = wave >> 2;     // 4 (M) x 2 (N) waves
  const int row0 = blockIdx.x * 128;
  const int col0 = blockIdx.y * 128;

  // each thread owns 2 A-chunks + 2 B-chunks of 8 bf16 per stage (4 async b128)
  const int c0 = tid,        r0c = c0 >> 2, k0c = (c0 & 3) << 3;
  const int c1 = tid + 256,  r1c = c1 >> 2, k1c = (c1 & 3) << 3;
  int ga0 = row0 + r0c; if (ga0 >= M) ga0 = M - 1;   // clamp: rows >= M never stored
  int ga1 = row0 + r1c; if (ga1 >= M) ga1 = M - 1;
  const u16* Ap0 = A  + (size_t)ga0 * K + k0c;
  const u16* Ap1 = A  + (size_t)ga1 * K + k1c;
  const u16* Bp0 = Bt + (size_t)(col0 + r0c) * K + k0c;
  const u16* Bp1 = Bt + (size_t)(col0 + r1c) * K + k1c;

  auto stage = [&](int s) {
    const int buf = s & 1;
    const int ko  = s << 5;
    copy16_to_lds(&As[buf][r0c * 32 + k0c], Ap0 + ko);
    copy16_to_lds(&As[buf][r1c * 32 + k1c], Ap1 + ko);
    copy16_to_lds(&Bs[buf][r0c * 32 + k0c], Bp0 + ko);
    copy16_to_lds(&Bs[buf][r1c * 32 + k1c], Bp1 + ko);
  };

  f32x8 acc[2][4];
  const f32x8 z = {0.f,0.f,0.f,0.f,0.f,0.f,0.f,0.f};
  for (int i = 0; i < 2; ++i)
    for (int j = 0; j < 4; ++j) acc[i][j] = z;

  const int nk = K >> 5;
  stage(0);
  for (int s = 0; s < nk; ++s) {
    if (s + 1 < nk) { stage(s + 1); wait_async<4>(); }  // stage s done (in-order)
    else            { wait_async<0>(); }
    __syncthreads();                                    // everyone's tile visible

    const u16* Ab = As[s & 1];
    const u16* Bb = Bs[s & 1];
    Frag a[2], b[4];
    // A frag (16x32 bf16): lanes 0-15 K0-7|K16-23; lanes 16-31 K8-15|K24-31
    for (int f = 0; f < 2; ++f) {
      int r = wm * 32 + f * 16 + lr;
      a[f].q[0] = *(const uint4*)(&Ab[r * 32 + lh * 8]);
      a[f].q[1] = *(const uint4*)(&Ab[r * 32 + 16 + lh * 8]);
    }
    // B frag (32x16 bf16): lanes 0-15 K0-7|K16-23; lanes 16-31 K8-15|K24-31 (N-major staged)
    for (int f = 0; f < 4; ++f) {
      int n = wn * 64 + f * 16 + lr;
      b[f].q[0] = *(const uint4*)(&Bb[n * 32 + lh * 8]);
      b[f].q[1] = *(const uint4*)(&Bb[n * 32 + 16 + lh * 8]);
    }
    for (int i = 0; i < 2; ++i)
      for (int j = 0; j < 4; ++j)
        acc[i][j] = __builtin_amdgcn_wmma_f32_16x16x32_bf16(
            false, a[i].v, false, b[j].v, (short)0, acc[i][j], false, false);

    __syncthreads();   // frags consumed before stage s+2 overwrites this buffer
  }

  // epilogue: C 16x16 layout -> VGPR g: M=g (lanes0-15) / M=8+g (lanes16-31), N=lr
  for (int i = 0; i < 2; ++i) {
    int rbase = row0 + wm * 32 + i * 16 + lh * 8;
    for (int j = 0; j < 4; ++j) {
      int col = col0 + wn * 64 + j * 16 + lr;
      float bv = bias ? bias[col] : 0.f;
      for (int g = 0; g < 8; ++g) {
        int row = rbase + g;
        if (row >= M) continue;
        float vv = acc[i][j][g] + bv;
        if (gelu) vv = 0.5f * vv * (1.0f + erff(vv * 0.70710678118f));
        if (res)  vv += res[(size_t)row * N + col];
        if (Cf)   Cf[(size_t)row * N + col] = vv;
        if (Cbf)  Cbf[(size_t)row * N + col] = f2bf(vv);
      }
    }
  }
}

// ---------- attention: wave per edge, 12 strided elems/lane, head = t/2 ----------
__global__ void k_edge_logits(const float* __restrict__ q, const float* __restrict__ k,
                              const u16* __restrict__ e,
                              const int* __restrict__ src, const int* __restrict__ dst,
                              float* __restrict__ logits, u32* __restrict__ mseg, int E) {
  long gt = (long)blockIdx.x * blockDim.x + threadIdx.x;
  int eid = (int)(gt >> 5);
  int lane = threadIdx.x & 31;
  if (eid >= E) return;
  int s = src[eid], d = dst[eid];
  const float* qp = q + (size_t)d * 384;
  const float* kp = k + (size_t)s * 384;
  const u16*   ep = e + (size_t)eid * 384;
  float acc[6] = {0.f,0.f,0.f,0.f,0.f,0.f};
  #pragma unroll
  for (int t = 0; t < 12; ++t) {
    int i = lane + t * 32;
    acc[t >> 1] += qp[i] * (kp[i] + bf2f(ep[i]));
  }
  #pragma unroll
  for (int h = 0; h < 6; ++h) {
    float v = acc[h];
    for (int o = 16; o > 0; o >>= 1) v += __shfl_xor(v, o, 32);
    acc[h] = v;
  }
  if (lane == 0) {
    #pragma unroll
    for (int h = 0; h < 6; ++h) {
      float lg = acc[h] * 0.125f;            // 1/sqrt(64)
      logits[(size_t)eid * 6 + h] = lg;
      atomicMax(&mseg[(size_t)d * 6 + h], f2ord(lg));
    }
  }
}

__global__ void k_edge_exp(float* __restrict__ logits, const int* __restrict__ dst,
                           const u32* __restrict__ mseg, float* __restrict__ ssum, long EH) {
  long i = (long)blockIdx.x * blockDim.x + threadIdx.x;
  if (i >= EH) return;
  int eid = (int)(i / 6), h = (int)(i % 6);
  int d = dst[eid];
  float m = ord2f(mseg[(size_t)d * 6 + h]);
  float a = expf(logits[i] - m);
  logits[i] = a;                              // reuse buffer for a
  atomicAdd(&ssum[(size_t)d * 6 + h], a);
}

__global__ void k_edge_agg(const float* __restrict__ a, const float* __restrict__ ssum,
                           const float* __restrict__ v, const u16* __restrict__ e,
                           const int* __restrict__ src, const int* __restrict__ dst,
                           float* __restrict__ agg, int E) {
  long gt = (long)blockIdx.x * blockDim.x + threadIdx.x;
  int eid = (int)(gt >> 5);
  int lane = threadIdx.x & 31;
  if (eid >= E) return;
  int s = src[eid], d = dst[eid];
  float coef[6];
  #pragma unroll
  for (int h = 0; h < 6; ++h) {
    float den = ssum[(size_t)d * 6 + h];
    coef[h] = a[(size_t)eid * 6 + h] / (den == 0.f ? 1.f : den);
  }
  const float* vp = v + (size_t)s * 384;
  const u16*   ep = e + (size_t)eid * 384;
  float* ap = agg + (size_t)d * 384;
  #pragma unroll
  for (int t = 0; t < 12; ++t) {
    int i = lane + t * 32;
    atomicAdd(&ap[i], coef[t >> 1] * (vp[i] + bf2f(ep[i])));
  }
}

__global__ void k_resid(const float* __restrict__ x, const float* __restrict__ agg,
                        const float* __restrict__ skip, float* __restrict__ x1, long n) {
  long i = (long)blockIdx.x * blockDim.x + threadIdx.x;
  if (i < n) x1[i] = x[i] + agg[i] + skip[i];
}

// ---------- host ----------
static inline unsigned cdiv(long a, long b) { return (unsigned)((a + b - 1) / b); }

extern "C" void kernel_launch(void* const* d_in, const int* in_sizes, int n_in,
                              void* d_out, int out_size, void* d_ws, size_t ws_size,
                              hipStream_t stream) {
  const float* x    = (const float*)d_in[0];
  const int*   ei   = (const int*)d_in[1];
  const float* ea   = (const float*)d_in[2];
  const float* g1   = (const float*)d_in[3];
  const float* b1   = (const float*)d_in[4];
  const float* Wq   = (const float*)d_in[5];
  const float* bq   = (const float*)d_in[6];
  const float* Wk   = (const float*)d_in[7];
  const float* bk   = (const float*)d_in[8];
  const float* Wv   = (const float*)d_in[9];
  const float* bv   = (const float*)d_in[10];
  const float* We   = (const float*)d_in[11];
  const float* Wsk  = (const float*)d_in[12];
  const float* bsk  = (const float*)d_in[13];
  const float* g2   = (const float*)d_in[14];
  const float* b2   = (const float*)d_in[15];
  const float* W1   = (const float*)d_in[16];
  const float* bm1  = (const float*)d_in[17];
  const float* W2   = (const float*)d_in[18];
  const float* bm2  = (const float*)d_in[19];

  const int D  = 384;
  const int N  = in_sizes[0] / D;
  const int E  = in_sizes[1] / 2;
  const int DH = in_sizes[17];           // 768
  const int H  = 6;
  const int* src = ei;
  const int* dst = ei + E;

  size_t off = 0;
  auto wsAlloc = [&](size_t bytes) {
    void* p = (char*)d_ws + off;
    off += (bytes + 255) & ~(size_t)255;
    return p;
  };
  u16*   hbf   = (u16*)  wsAlloc((size_t)N * D * 2);
  u16*   abf   = (u16*)  wsAlloc((size_t)E * D * 2);
  u16*   WqT   = (u16*)  wsAlloc((size_t)D * D * 2);
  u16*   WkT   = (u16*)  wsAlloc((size_t)D * D * 2);
  u16*   WvT   = (u16*)  wsAlloc((size_t)D * D * 2);
  u16*   WeT   = (u16*)  wsAlloc((size_t)D * D * 2);
  u16*   WskT  = (u16*)  wsAlloc((size_t)D * D * 2);
  u16*   W1T   = (u16*)  wsAlloc((size_t)D * DH * 2);
  u16*   W2T   = (u16*)  wsAlloc((size_t)DH * D * 2);
  float* qb    = (float*)wsAlloc((size_t)N * D * 4);
  float* kb    = (float*)wsAlloc((size_t)N * D * 4);
  float* vb    = (float*)wsAlloc((size_t)N * D * 4);
  float* skipb = (float*)wsAlloc((size_t)N * D * 4);
  u16*   ebf   = (u16*)  wsAlloc((size_t)E * D * 2);
  float* logit = (float*)wsAlloc((size_t)E * H * 4);
  u32*   mseg  = (u32*)  wsAlloc((size_t)N * H * 4);
  float* ssum  = (float*)wsAlloc((size_t)N * H * 4);
  float* agg   = (float*)wsAlloc((size_t)N * D * 4);
  float* x1    = (float*)wsAlloc((size_t)N * D * 4);
  u16*   h2bf  = (u16*)  wsAlloc((size_t)N * D * 2);
  u16*   hidbf = (u16*)  wsAlloc((size_t)N * DH * 2);
  float* bnsum = (float*)wsAlloc((size_t)2 * D * 4);   // sum | sumsq
  float* bnsc  = (float*)wsAlloc((size_t)2 * D * 4);   // scale | shift
  float* bnss  = bnsum + D;
  float* bnsh  = bnsc + D;

  const long ND = (long)N * D;
  const long ED = (long)E * D;

  // ---- BN1 -> hbf (bf16) ----
  k_zero<<<cdiv(2 * D, 256), 256, 0, stream>>>((u32*)bnsum, 2 * D);
  k_bn_stats<<<cdiv(N, 256), D, 0, stream>>>(x, bnsum, bnss, N, D);
  k_bn_finalize<<<cdiv(D, 256), 256, 0, stream>>>(bnsum, bnss, g1, b1, bnsc, bnsh, N, D);
  k_bn_apply<<<cdiv(ND, 256), 256, 0, stream>>>(x, bnsc, bnsh, hbf, ND, D);

  // ---- conversions ----
  k_f2bf<<<cdiv(ED, 256), 256, 0, stream>>>(ea, abf, ED);
  k_wtrans<<<cdiv((long)D * D, 256), 256, 0, stream>>>(Wq,  WqT,  D, D);
  k_wtrans<<<cdiv((long)D * D, 256), 256, 0, stream>>>(Wk,  WkT,  D, D);
  k_wtrans<<<cdiv((long)D * D, 256), 256, 0, stream>>>(Wv,  WvT,  D, D);
  k_wtrans<<<cdiv((long)D * D, 256), 256, 0, stream>>>(We,  WeT,  D, D);
  k_wtrans<<<cdiv((long)D * D, 256), 256, 0, stream>>>(Wsk, WskT, D, D);
  k_wtrans<<<cdiv((long)D * DH, 256), 256, 0, stream>>>(W1, W1T, D, DH);
  k_wtrans<<<cdiv((long)DH * D, 256), 256, 0, stream>>>(W2, W2T, DH, D);

  // ---- projections (WMMA) ----
  dim3 gN(cdiv(N, 128), D / 128);
  k_gemm<<<gN, 256, 0, stream>>>(hbf, WqT,  bq,  nullptr, qb,    nullptr, N, D, D, 0);
  k_gemm<<<gN, 256, 0, stream>>>(hbf, WkT,  bk,  nullptr, kb,    nullptr, N, D, D, 0);
  k_gemm<<<gN, 256, 0, stream>>>(hbf, WvT,  bv,  nullptr, vb,    nullptr, N, D, D, 0);
  k_gemm<<<gN, 256, 0, stream>>>(hbf, WskT, bsk, nullptr, skipb, nullptr, N, D, D, 0);
  dim3 gE(cdiv(E, 128), D / 128);
  k_gemm<<<gE, 256, 0, stream>>>(abf, WeT, nullptr, nullptr, nullptr, ebf, E, D, D, 0);

  // ---- segment-softmax attention ----
  k_zero<<<cdiv((long)N * H, 256), 256, 0, stream>>>(mseg, (long)N * H);
  k_zero<<<cdiv((long)N * H, 256), 256, 0, stream>>>((u32*)ssum, (long)N * H);
  k_zero<<<cdiv(ND, 256), 256, 0, stream>>>((u32*)agg, ND);
  k_edge_logits<<<cdiv((long)E * 32, 256), 256, 0, stream>>>(qb, kb, ebf, src, dst, logit, mseg, E);
  k_edge_exp<<<cdiv((long)E * H, 256), 256, 0, stream>>>(logit, dst, mseg, ssum, (long)E * H);
  k_edge_agg<<<cdiv((long)E * 32, 256), 256, 0, stream>>>(logit, ssum, vb, ebf, src, dst, agg, E);
  k_resid<<<cdiv(ND, 256), 256, 0, stream>>>(x, agg, skipb, x1, ND);

  // ---- BN2 -> h2bf ----
  k_zero<<<cdiv(2 * D, 256), 256, 0, stream>>>((u32*)bnsum, 2 * D);
  k_bn_stats<<<cdiv(N, 256), D, 0, stream>>>(x1, bnsum, bnss, N, D);
  k_bn_finalize<<<cdiv(D, 256), 256, 0, stream>>>(bnsum, bnss, g2, b2, bnsc, bnsh, N, D);
  k_bn_apply<<<cdiv(ND, 256), 256, 0, stream>>>(x1, bnsc, bnsh, h2bf, ND, D);

  // ---- MLP (WMMA, fused GELU + residual) ----
  dim3 gM1(cdiv(N, 128), DH / 128);
  k_gemm<<<gM1, 256, 0, stream>>>(h2bf, W1T, bm1, nullptr, nullptr, hidbf, N, DH, D, 1);
  dim3 gM2(cdiv(N, 128), D / 128);
  k_gemm<<<gM2, 256, 0, stream>>>(hidbf, W2T, bm2, x1, (float*)d_out, nullptr, N, D, DH, 0);
}
